// MoE_14456859918466
// MI455X (gfx1250) — matrix operand order repaired and checked by reading, not compiled, not run on hardware
//
#include <hip/hip_runtime.h>

#define S_TOK 16384
#define M_DIM 512
#define E_NUM 16
#define K_TOP 2
#define D_DIM 256
#define H1R 500
#define H2R 500
#define H3R 1000
#define R_MAX (S_TOK * K_TOP + 32 * E_NUM)   // 33280, multiple of 32

typedef __attribute__((ext_vector_type(16))) __bf16 v16bf;
typedef __attribute__((ext_vector_type(8)))  float  v8f;

struct __align__(16) U4 { unsigned int a, b, c, d; };
union ABfrag { v16bf v; U4 q[2]; };

__device__ __forceinline__ unsigned short f2bf(float f) {
    unsigned int u = __float_as_uint(f);
    unsigned int r = u + 0x7FFFu + ((u >> 16) & 1u);   // round-to-nearest-even
    return (unsigned short)(r >> 16);
}

// ---------------- init: zero stats, fill token_of_row with -1 ----------------
__global__ void init_kernel(float* proxy, int* cnt, int* cursor, int* tok) {
    int i = blockIdx.x * blockDim.x + threadIdx.x;
    if (i < E_NUM) { proxy[i] = 0.f; cnt[i] = 0; cursor[i] = 0; }
    if (i < R_MAX) tok[i] = -1;
}

// ---------------- router: logits, softmax, top-2, stats ----------------
__global__ __launch_bounds__(256)
void router_kernel(const float* __restrict__ x, const float* __restrict__ noise,
                   const float* __restrict__ Wr, float* __restrict__ select0,
                   float* __restrict__ gate, int* __restrict__ idx,
                   float* __restrict__ proxy, int* __restrict__ cnt) {
    __shared__ float sWr[E_NUM * M_DIM];
    __shared__ float sprox[E_NUM];
    __shared__ int   scnt[E_NUM];
    int tid = threadIdx.x;
    for (int i = tid; i < E_NUM * M_DIM; i += 256) sWr[i] = Wr[i];
    if (tid < E_NUM) { sprox[tid] = 0.f; scnt[tid] = 0; }
    __syncthreads();

    int s = blockIdx.x * 256 + tid;
    float acc[E_NUM];
#pragma unroll
    for (int e = 0; e < E_NUM; ++e) acc[e] = 0.f;
    const float* xr = x + (size_t)s * M_DIM;
    for (int m = 0; m < M_DIM; ++m) {
        float xv = xr[m];
#pragma unroll
        for (int e = 0; e < E_NUM; ++e) acc[e] += xv * sWr[e * M_DIM + m];
    }
#pragma unroll
    for (int e = 0; e < E_NUM; ++e) acc[e] += noise[(size_t)s * E_NUM + e];

    float mx = acc[0];
#pragma unroll
    for (int e = 1; e < E_NUM; ++e) mx = fmaxf(mx, acc[e]);
    float sel[E_NUM]; float sum = 0.f;
#pragma unroll
    for (int e = 0; e < E_NUM; ++e) { sel[e] = __expf(acc[e] - mx); sum += sel[e]; }
    float inv = 1.f / sum;
#pragma unroll
    for (int e = 0; e < E_NUM; ++e) sel[e] *= inv;

    int i0 = 0; float g0 = sel[0];
#pragma unroll
    for (int e = 1; e < E_NUM; ++e) if (sel[e] > g0) { g0 = sel[e]; i0 = e; }
    int i1 = -1; float g1 = -1.f;
#pragma unroll
    for (int e = 0; e < E_NUM; ++e) if (e != i0 && sel[e] > g1) { g1 = sel[e]; i1 = e; }

    float gs = 1.f / (g0 + g1);
    gate[s * 2 + 0] = g0 * gs; gate[s * 2 + 1] = g1 * gs;
    idx[s * 2 + 0] = i0;       idx[s * 2 + 1] = i1;
#pragma unroll
    for (int e = 0; e < E_NUM; ++e)
        select0[(size_t)s * E_NUM + e] = (e == i0 || e == i1) ? 1.f : 0.f;

#pragma unroll
    for (int e = 0; e < E_NUM; ++e) atomicAdd(&sprox[e], sel[e]);
    atomicAdd(&scnt[i0], 1); atomicAdd(&scnt[i1], 1);
    __syncthreads();
    if (tid < E_NUM) { atomicAdd(&proxy[tid], sprox[tid]); atomicAdd(&cnt[tid], scnt[tid]); }
}

// ---------------- scan: 32-aligned segment offsets + balance loss ----------------
__global__ void scan_kernel(const float* proxy, const int* cnt, int* offsets, float* out_tail) {
    if (threadIdx.x == 0 && blockIdx.x == 0) {
        int off = 0; float bal = 0.f;
        for (int e = 0; e < E_NUM; ++e) {
            offsets[e] = off;
            off += (cnt[e] + 31) & ~31;
            bal += (proxy[e] / (float)S_TOK) * ((float)cnt[e] / (float)S_TOK);
        }
        offsets[E_NUM] = off;
        out_tail[0] = bal * (float)E_NUM;   // mean_e(p*d)*E^2 == E*sum
        out_tail[1] = 0.f;                  // dist_loss
    }
}

// ---------------- assign rows ----------------
__global__ void assign_kernel(const int* __restrict__ idx, const int* __restrict__ offsets,
                              int* cursor, int* __restrict__ tok, int* __restrict__ row_of) {
    int i = blockIdx.x * blockDim.x + threadIdx.x;
    if (i >= S_TOK * K_TOP) return;
    int e = idx[i];
    int r = offsets[e] + atomicAdd(&cursor[e], 1);
    tok[r] = i / K_TOP;
    row_of[i] = r;
}

// ---------------- gather x -> bf16 rows ----------------
__global__ void gather_kernel(const float* __restrict__ x, const int* __restrict__ tok,
                              unsigned short* __restrict__ xg) {
    long i = (long)blockIdx.x * blockDim.x + threadIdx.x;
    if (i >= (long)R_MAX * M_DIM) return;
    int row = (int)(i >> 9);
    int m = (int)(i & 511);
    int t = tok[row];
    float v = (t >= 0) ? x[(size_t)t * M_DIM + m] : 0.f;
    xg[i] = f2bf(v);
}

// ---------------- weight swizzle: f32 [E][KREAL][NREAL] -> bf16 WMMA B-fragments ----------------
// layout: dword index (((e*NT + nt)*KT + kt)*32 + lane)*8 + j
// B frag element mapping: n = nt*16 + (lane&15), k0 = kt*32 + (lane>=16?16:0) + 2*j
template<int KREAL, int NREAL, int KPAD, int NPAD>
__global__ void swizzle_kernel(const float* __restrict__ W, unsigned int* __restrict__ out) {
    constexpr int NT = NPAD / 16, KT = KPAD / 32;
    constexpr int total = E_NUM * NT * KT * 256;
    int i = blockIdx.x * blockDim.x + threadIdx.x;
    if (i >= total) return;
    int j    = i & 7;
    int lane = (i >> 3) & 31;
    int kt   = (i >> 8) % KT;
    int nt   = (i / (256 * KT)) % NT;
    int e    = i / (256 * KT * NT);
    int n  = nt * 16 + (lane & 15);
    int k0 = kt * 32 + ((lane >> 4) & 1) * 16 + j * 2;
    float f0 = (k0     < KREAL && n < NREAL) ? W[((size_t)e * KREAL + k0    ) * NREAL + n] : 0.f;
    float f1 = (k0 + 1 < KREAL && n < NREAL) ? W[((size_t)e * KREAL + k0 + 1) * NREAL + n] : 0.f;
    out[i] = (unsigned int)f2bf(f0) | ((unsigned int)f2bf(f1) << 16);
}

// ---------------- expert-segmented WMMA GEMM ----------------
// block = 128 threads (4 waves); each wave: 32 rows x 64 cols (2 A frags reuse 4 B frags);
// block covers 256 cols.
template<int KPAD, int NPAD, bool RELU, bool F32OUT>
__global__ __launch_bounds__(128)
void expert_gemm(const unsigned short* __restrict__ in,   // bf16 [R_MAX][KPAD]
                 const unsigned short* __restrict__ wsw,  // swizzled bf16 B fragments
                 const float* __restrict__ bias, int nreal,
                 void* __restrict__ outv,
                 const int* __restrict__ offsets) {
    constexpr int NT = NPAD / 16, KT = KPAD / 32;
    const int lane = threadIdx.x & 31;
    const int wv   = threadIdx.x >> 5;
    const int rowBase = blockIdx.x * 32;
    const int colBase = blockIdx.y * 256 + wv * 64;

    int e = 0;
#pragma unroll
    for (int t = 1; t < E_NUM; ++t) e = (rowBase >= offsets[t]) ? t : e;

    const int h = lane >> 4;
    const unsigned short* ap0 = in + (size_t)(rowBase + (lane & 15)) * KPAD;
    const unsigned short* ap1 = ap0 + (size_t)16 * KPAD;
    const int nt0 = colBase >> 4;
    const unsigned short* wb = wsw + ((size_t)e * NT + nt0) * (size_t)KT * 512 + (size_t)lane * 16;

    v8f z = {0.f, 0.f, 0.f, 0.f, 0.f, 0.f, 0.f, 0.f};
    v8f cc0[4] = {z, z, z, z};
    v8f cc1[4] = {z, z, z, z};

    for (int kt = 0; kt < KT; ++kt) {
        ABfrag a0, a1;
        const U4* a0q = (const U4*)(ap0 + kt * 32 + h * 16);
        a0.q[0] = a0q[0];    // K = [h*16, h*16+7]
        a0.q[1] = a0q[1];    // K = [h*16+8, h*16+15]
        const U4* a1q = (const U4*)(ap1 + kt * 32 + h * 16);
        a1.q[0] = a1q[0];
        a1.q[1] = a1q[1];

        ABfrag bf[4];
        const unsigned short* wkt = wb + (size_t)kt * 512;
#pragma unroll
        for (int t = 0; t < 4; ++t) {
            const U4* bp = (const U4*)(wkt + (size_t)t * KT * 512);
            bf[t].q[0] = bp[0];
            bf[t].q[1] = bp[1];
        }
#pragma unroll
        for (int t = 0; t < 4; ++t) {
            cc0[t] = __builtin_amdgcn_wmma_f32_16x16x32_bf16(
                false, a0.v, false, bf[t].v, (short)0, cc0[t], false, false);
            cc1[t] = __builtin_amdgcn_wmma_f32_16x16x32_bf16(
                false, a1.v, false, bf[t].v, (short)0, cc1[t], false, false);
        }
    }

    float* fo = (float*)outv;
    unsigned short* bo = (unsigned short*)outv;
#pragma unroll
    for (int t = 0; t < 4; ++t) {
        int n = colBase + t * 16 + (lane & 15);
        float badd = (n < nreal) ? bias[(size_t)e * nreal + n] : 0.f;
#pragma unroll
        for (int m = 0; m < 2; ++m) {
#pragma unroll
            for (int r = 0; r < 8; ++r) {
                float v = (m == 0 ? cc0[t][r] : cc1[t][r]) + badd;
                if (RELU) v = v > 0.f ? v : 0.f;
                size_t oi = (size_t)(rowBase + m * 16 + h * 8 + r) * NPAD + n;
                if (F32OUT) fo[oi] = v; else bo[oi] = f2bf(v);
            }
        }
    }
}

// ---------------- combine: final_output[s,d] = g0*row0 + g1*row1 ----------------
__global__ void combine_kernel(const float* __restrict__ out4, const float* __restrict__ gate,
                               const int* __restrict__ row_of, float* __restrict__ out) {
    int i = blockIdx.x * blockDim.x + threadIdx.x;
    if (i >= S_TOK * D_DIM) return;
    int s = i >> 8, d = i & 255;
    int r0 = row_of[s * 2 + 0], r1 = row_of[s * 2 + 1];
    out[i] = gate[s * 2 + 0] * out4[(size_t)r0 * D_DIM + d]
           + gate[s * 2 + 1] * out4[(size_t)r1 * D_DIM + d];
}

extern "C" void kernel_launch(void* const* d_in, const int* in_sizes, int n_in,
                              void* d_out, int out_size, void* d_ws, size_t ws_size,
                              hipStream_t stream) {
    const float* x     = (const float*)d_in[0];
    const float* noise = (const float*)d_in[1];
    const float* Wr    = (const float*)d_in[2];
    const float* W1    = (const float*)d_in[3];
    const float* b1    = (const float*)d_in[4];
    const float* W2    = (const float*)d_in[5];
    const float* b2    = (const float*)d_in[6];
    const float* W3    = (const float*)d_in[7];
    const float* b3    = (const float*)d_in[8];
    const float* W4    = (const float*)d_in[9];
    const float* b4    = (const float*)d_in[10];

    float* out_final = (float*)d_out;                          // [S,D]
    float* out_sel   = out_final + (size_t)S_TOK * D_DIM;      // [S,E]
    float* out_tail  = out_sel + (size_t)S_TOK * E_NUM;        // balance, dist

    // ---- workspace layout (256B aligned) ----
    char* ws = (char*)d_ws;
    size_t o = 0;
    auto alloc = [&](size_t bytes) { size_t r = o; o = (o + bytes + 255) & ~(size_t)255; return r; };
    float* gate    = (float*)(ws + alloc((size_t)S_TOK * K_TOP * 4));
    int*   idx     = (int*)  (ws + alloc((size_t)S_TOK * K_TOP * 4));
    int*   row_of  = (int*)  (ws + alloc((size_t)S_TOK * K_TOP * 4));
    int*   tok     = (int*)  (ws + alloc((size_t)R_MAX * 4));
    float* proxy   = (float*)(ws + alloc(E_NUM * 4));
    int*   cnt     = (int*)  (ws + alloc(E_NUM * 4));
    int*   cursor  = (int*)  (ws + alloc(E_NUM * 4));
    int*   offsets = (int*)  (ws + alloc((E_NUM + 1) * 4));
    // region A: xg + h1, later reused for h3 (same total size)
    size_t XG_B = (size_t)R_MAX * 512 * 2;
    size_t regA = alloc(2 * XG_B);
    unsigned short* xg = (unsigned short*)(ws + regA);
    unsigned short* h1 = (unsigned short*)(ws + regA + XG_B);
    unsigned short* h3 = (unsigned short*)(ws + regA);           // overlays dead xg+h1
    // region B: h2, later reused for out4 (same size)
    size_t regB = alloc((size_t)R_MAX * 512 * 2);
    unsigned short* h2 = (unsigned short*)(ws + regB);
    float* out4        = (float*)(ws + regB);                    // overlays dead h2
    unsigned short* w1sw = (unsigned short*)(ws + alloc((size_t)E_NUM * 512 * 512 * 2));
    unsigned short* w2sw = (unsigned short*)(ws + alloc((size_t)E_NUM * 512 * 512 * 2));
    unsigned short* w3sw = (unsigned short*)(ws + alloc((size_t)E_NUM * 512 * 1024 * 2));
    unsigned short* w4sw = (unsigned short*)(ws + alloc((size_t)E_NUM * 1024 * 256 * 2));
    (void)ws_size; (void)n_in; (void)in_sizes; (void)out_size;

    // ---- pipeline ----
    init_kernel<<<(R_MAX + 255) / 256, 256, 0, stream>>>(proxy, cnt, cursor, tok);

    router_kernel<<<S_TOK / 256, 256, 0, stream>>>(x, noise, Wr, out_sel, gate, idx, proxy, cnt);

    scan_kernel<<<1, 1, 0, stream>>>(proxy, cnt, offsets, out_tail);

    assign_kernel<<<(S_TOK * K_TOP + 255) / 256, 256, 0, stream>>>(idx, offsets, cursor, tok, row_of);

    gather_kernel<<<(int)(((size_t)R_MAX * M_DIM + 255) / 256), 256, 0, stream>>>(x, tok, xg);

    swizzle_kernel<512, 500, 512, 512><<<(E_NUM * 32 * 16 * 256) / 256, 256, 0, stream>>>(W1, (unsigned int*)w1sw);
    swizzle_kernel<500, 500, 512, 512><<<(E_NUM * 32 * 16 * 256) / 256, 256, 0, stream>>>(W2, (unsigned int*)w2sw);
    swizzle_kernel<500, 1000, 512, 1024><<<(E_NUM * 64 * 16 * 256) / 256, 256, 0, stream>>>(W3, (unsigned int*)w3sw);
    swizzle_kernel<1000, 256, 1024, 256><<<(E_NUM * 16 * 32 * 256) / 256, 256, 0, stream>>>(W4, (unsigned int*)w4sw);

    dim3 blk(128, 1, 1);
    expert_gemm<512, 512, true, false><<<dim3(R_MAX / 32, 2), blk, 0, stream>>>(xg, w1sw, b1, H1R, (void*)h1, offsets);
    expert_gemm<512, 512, true, false><<<dim3(R_MAX / 32, 2), blk, 0, stream>>>(h1, w2sw, b2, H2R, (void*)h2, offsets);
    expert_gemm<512, 1024, true, false><<<dim3(R_MAX / 32, 4), blk, 0, stream>>>(h2, w3sw, b3, H3R, (void*)h3, offsets);
    expert_gemm<1024, 256, false, true><<<dim3(R_MAX / 32, 1), blk, 0, stream>>>(h3, w4sw, b4, D_DIM, (void*)out4, offsets);

    combine_kernel<<<(S_TOK * D_DIM) / 256, 256, 0, stream>>>(out4, gate, row_of, out_final);
}
